// TracktentionLayer_63488206570031
// MI455X (gfx1250) — matrix operand order repaired and checked by reading, not compiled, not run on hardware
//
#include <hip/hip_runtime.h>
#include <hip/hip_bf16.h>
#include <stdint.h>

typedef __attribute__((ext_vector_type(16))) _Float16 v16h;
typedef __attribute__((ext_vector_type(8)))  _Float16 v8h;
typedef __attribute__((ext_vector_type(8)))  float    v8f;

#define T_DIM   16
#define HW_N    729
#define HW_P    736      // padded to 46*16
#define M_TRK   512
#define D_M     1152
#define H_N     8
#define HD      144
#define HD_P    160      // padded to 5*32
#define QUARTER 288
#define HALF    576

static __device__ __forceinline__ v16h cat16(v8h lo, v8h hi) {
  return __builtin_shufflevector(lo, hi, 0,1,2,3,4,5,6,7,8,9,10,11,12,13,14,15);
}
static __device__ __forceinline__ v8f wmma_f16(v16h a, v16h b, v8f c) {
  return __builtin_amdgcn_wmma_f32_16x16x32_f16(false, a, false, b, (short)0, c, false, false);
}
static __device__ __forceinline__ float wred_sum(float v) {
#pragma unroll
  for (int o = 16; o > 0; o >>= 1) v += __shfl_xor(v, o, 32);
  return v;
}
static __device__ __forceinline__ float wred_max(float v) {
#pragma unroll
  for (int o = 16; o > 0; o >>= 1) v = fmaxf(v, __shfl_xor(v, o, 32));
  return v;
}

// ---------------- fp32 -> f16 convert ----------------
__global__ void k_f32_to_f16(const float* __restrict__ in, _Float16* __restrict__ out, int n) {
  int i = blockIdx.x * 256 + threadIdx.x;
  if (i < n) out[i] = (_Float16)in[i];
}

// ---------------- zero fill (16B granules) ----------------
__global__ void k_zero(uint4* __restrict__ p, size_t n16) {
  size_t i = (size_t)blockIdx.x * 256 + threadIdx.x;
  size_t stride = (size_t)gridDim.x * 256;
  uint4 z; z.x = 0; z.y = 0; z.z = 0; z.w = 0;
  for (; i < n16; i += stride) p[i] = z;
}

// ---------------- register-blocked WMMA GEMM: C[M,N] = A[M,K] * W[N,K]^T ----------
// A, W: f16 row-major (K contiguous). 8 waves/block arranged 4x2; each wave owns a
// 64x32 C region = 4x2 WMMA tiles (8 accumulators). A-frag reused 2x, B-frag 4x.
__global__ __launch_bounds__(256) void k_gemm_f16f16_f32(
    const _Float16* __restrict__ A, const _Float16* __restrict__ W,
    float* __restrict__ C, int Mrows, int Ncols, int Kdim)
{
  const int wave = threadIdx.x >> 5;
  const int lane = threadIdx.x & 31;
  const int g  = lane >> 4;
  const int nl = lane & 15;
  const int wrow = wave & 3;                       // 0..3
  const int wcol = wave >> 2;                      // 0..1
  const int mBase = blockIdx.y * 256 + wrow * 64;  // wave's 64-row strip
  const int nBase = blockIdx.x * 64 + wcol * 32;   // wave's 32-col strip

  // A row pointers; clamp OOB tail rows (results discarded at store)
  const _Float16* pa[4];
#pragma unroll
  for (int r = 0; r < 4; ++r) {
    int row = mBase + r * 16 + nl;
    if (row > Mrows - 1) row = Mrows - 1;
    pa[r] = A + (size_t)row * Kdim;
  }
  const _Float16* pb0 = W + (size_t)(nBase + nl) * Kdim;
  const _Float16* pb1 = W + (size_t)(nBase + 16 + nl) * Kdim;

  const v8f zero = {};
  v8f c[4][2];
#pragma unroll
  for (int r = 0; r < 4; ++r) { c[r][0] = zero; c[r][1] = zero; }

  for (int k0 = 0; k0 < Kdim; k0 += 32) {
    v16h b0 = cat16(*(const v8h*)(pb0 + k0 + g * 16),
                    *(const v8h*)(pb0 + k0 + g * 16 + 8));
    v16h b1 = cat16(*(const v8h*)(pb1 + k0 + g * 16),
                    *(const v8h*)(pb1 + k0 + g * 16 + 8));
#pragma unroll
    for (int r = 0; r < 4; ++r) {
      v16h a = cat16(*(const v8h*)(pa[r] + k0 + g * 16),
                     *(const v8h*)(pa[r] + k0 + g * 16 + 8));
      c[r][0] = wmma_f16(a, b0, c[r][0]);
      c[r][1] = wmma_f16(a, b1, c[r][1]);
    }
  }
#pragma unroll
  for (int r = 0; r < 4; ++r) {
    const int m0 = mBase + r * 16;
    if (m0 + 16 <= Mrows) {                         // wave-uniform guard
      float* pc = C + (size_t)(m0 + g * 8) * Ncols + nBase + nl;
#pragma unroll
      for (int i = 0; i < 8; ++i) {
        pc[(size_t)i * Ncols]      = c[r][0][i];
        pc[(size_t)i * Ncols + 16] = c[r][1][i];
      }
    }
  }
}

// ---------------- LayerNorm(Q) -> head-split padded f16 ----------------
__global__ __launch_bounds__(256) void k_ln_q(const float* __restrict__ Qraw,
                                              const float* __restrict__ g_q,
                                              _Float16* __restrict__ Qh)
{
  const int wave = threadIdx.x >> 5, lane = threadIdx.x & 31;
  const int row = blockIdx.x * 8 + wave;    // 0..8191
  const int t = row >> 9, m = row & 511;
  const float* p = Qraw + (size_t)row * D_M;
  float v[36]; float sum = 0.f, sq = 0.f;
#pragma unroll
  for (int i = 0; i < 36; ++i) { float x = p[lane + 32 * i]; v[i] = x; sum += x; sq += x * x; }
  sum = wred_sum(sum); sq = wred_sum(sq);
  const float mean = sum * (1.0f / D_M);
  const float inv = rsqrtf(sq * (1.0f / D_M) - mean * mean + 1e-6f);
#pragma unroll
  for (int i = 0; i < 36; ++i) {
    int d = lane + 32 * i;
    float y = (v[i] - mean) * inv * g_q[d];
    int h = d / HD, dd = d % HD;
    Qh[((size_t)(t * H_N + h) * M_TRK + m) * HD_P + dd] = (_Float16)y;
  }
}

// ---------------- RoPE + LayerNorm (K) -> head-split padded f16 ----------------
__global__ __launch_bounds__(256) void k_rope_ln_k(const float* __restrict__ Kraw,
                                                   const float* __restrict__ g_k,
                                                   _Float16* __restrict__ Kh)
{
  const int wave = threadIdx.x >> 5, lane = threadIdx.x & 31;
  const int row = blockIdx.x * 8 + wave;    // 0..11663
  if (row >= T_DIM * HW_N) return;
  const int t = row / HW_N, n = row % HW_N;
  const float px = (float)(n % 27), py = (float)(n / 27);
  const float* p = Kraw + (size_t)row * D_M;
  const float LQ = 4.6051702f / 288.0f;     // ln(100)/quarter
  float y[36]; float sum = 0.f, sq = 0.f;
#pragma unroll
  for (int i = 0; i < 18; ++i) {
    int pr = lane + 32 * i;                 // pair index 0..575
    int ti   = (pr < QUARTER) ? pr : pr - QUARTER;
    int base = (pr < QUARTER) ? 2 * pr : HALF + 2 * (pr - QUARTER);
    float pos = (pr < QUARTER) ? px : py;
    float ang = pos * __expf(-(float)ti * LQ);
    float cs = __cosf(ang), sn = __sinf(ang);
    float x1 = p[base], x2 = p[base + 1];
    float y1 = x1 * cs - x2 * sn, y2 = x1 * sn + x2 * cs;
    y[2 * i] = y1; y[2 * i + 1] = y2;
    sum += y1 + y2; sq += y1 * y1 + y2 * y2;
  }
  sum = wred_sum(sum); sq = wred_sum(sq);
  const float mean = sum * (1.0f / D_M);
  const float inv = rsqrtf(sq * (1.0f / D_M) - mean * mean + 1e-6f);
#pragma unroll
  for (int i = 0; i < 18; ++i) {
    int pr = lane + 32 * i;
    int base = (pr < QUARTER) ? 2 * pr : HALF + 2 * (pr - QUARTER);
#pragma unroll
    for (int q = 0; q < 2; ++q) {
      int d = base + q;
      float o = (y[2 * i + q] - mean) * inv * g_k[d];
      int h = d / HD, dd = d % HD;
      Kh[((size_t)(t * H_N + h) * HW_P + n) * HD_P + dd] = (_Float16)o;
    }
  }
}

// ---------------- RoPE(V=features) -> transposed f16 [T,H,HD,HW_P] ----------------
__global__ __launch_bounds__(256) void k_rope_v(const float* __restrict__ F,
                                                _Float16* __restrict__ Vt)
{
  const int wave = threadIdx.x >> 5, lane = threadIdx.x & 31;
  const int row = blockIdx.x * 8 + wave;
  if (row >= T_DIM * HW_N) return;
  const int t = row / HW_N, n = row % HW_N;
  const float px = (float)(n % 27), py = (float)(n / 27);
  const float* p = F + (size_t)row * D_M;
  const float LQ = 4.6051702f / 288.0f;
#pragma unroll
  for (int i = 0; i < 18; ++i) {
    int pr = lane + 32 * i;
    int ti   = (pr < QUARTER) ? pr : pr - QUARTER;
    int base = (pr < QUARTER) ? 2 * pr : HALF + 2 * (pr - QUARTER);
    float pos = (pr < QUARTER) ? px : py;
    float ang = pos * __expf(-(float)ti * LQ);
    float cs = __cosf(ang), sn = __sinf(ang);
    float x1 = p[base], x2 = p[base + 1];
    float y1 = x1 * cs - x2 * sn, y2 = x1 * sn + x2 * cs;
#pragma unroll
    for (int q = 0; q < 2; ++q) {
      int d = base + q;
      int h = d / HD, dd = d % HD;
      Vt[((size_t)(t * H_N + h) * HD + dd) * HW_P + n] = (_Float16)(q ? y2 : y1);
    }
  }
}

// ---------------- attention: one (t,h,16-query tile) per block ----------------
__global__ __launch_bounds__(256) void k_attn(
    const _Float16* __restrict__ Qh, const _Float16* __restrict__ Kh,
    const _Float16* __restrict__ Vt, const float* __restrict__ tracks,
    _Float16* __restrict__ sampled)
{
  __shared__ float S[16 * HW_P];            // 47104 B
  const int t = blockIdx.z, h = blockIdx.y, m0 = blockIdx.x * 16;
  const int wave = threadIdx.x >> 5, lane = threadIdx.x & 31;
  const int g = lane >> 4, nl = lane & 15;
  const _Float16* Qb = Qh + ((size_t)(t * H_N + h) * M_TRK + m0) * HD_P;
  const _Float16* Kb = Kh + (size_t)(t * H_N + h) * HW_P * HD_P;
  const v8f zero = {};

  // hoist the 5 Q A-fragments (loop-invariant across all 46 score tiles)
  v16h aQ[5];
#pragma unroll
  for (int kt = 0; kt < HD_P / 32; ++kt) {
    const _Float16* pa = Qb + nl * HD_P + kt * 32;
    aQ[kt] = cat16(*(const v8h*)(pa + g * 16), *(const v8h*)(pa + g * 16 + 8));
  }
  // hoist this lane's 8 track coordinates
  float tx[8], ty[8];
#pragma unroll
  for (int r = 0; r < 8; ++r) {
    const size_t mi = (size_t)t * M_TRK + m0 + g * 8 + r;
    tx[r] = tracks[mi * 2 + 0];
    ty[r] = tracks[mi * 2 + 1];
  }

  // ---- phase 1: scores = QK^T/12 + gaussian bias ----
  for (int j = wave; j < HW_P / 16; j += 8) {
    const int n0 = j * 16;
    v8f c = zero;
#pragma unroll
    for (int kt = 0; kt < HD_P / 32; ++kt) {
      const _Float16* pb = Kb + (size_t)(n0 + nl) * HD_P + kt * 32 + g * 16;
      c = wmma_f16(aQ[kt], cat16(*(const v8h*)pb, *(const v8h*)(pb + 8)), c);
    }
    const int nglob = n0 + nl;
    const float fx = (float)(nglob % 27), fy = (float)(nglob / 27);
#pragma unroll
    for (int r = 0; r < 8; ++r) {
      float s;
      if (nglob >= HW_N) s = -1.0e30f;
      else {
        const float dx = tx[r] - fx, dy = ty[r] - fy;
        s = c[r] * (1.0f / 12.0f) - 2.0f * (dx * dx + dy * dy);
      }
      S[(g * 8 + r) * HW_P + nglob] = s;
    }
  }
  __syncthreads();

  // ---- phase 2: softmax per row (wave-parallel) ----
  for (int row = wave; row < 16; row += 8) {
    float mx = -3.0e38f;
    for (int i = lane; i < HW_P; i += 32) mx = fmaxf(mx, S[row * HW_P + i]);
    mx = wred_max(mx);
    float sum = 0.f;
    for (int i = lane; i < HW_P; i += 32) {
      float pv = __expf(S[row * HW_P + i] - mx);
      S[row * HW_P + i] = pv; sum += pv;
    }
    sum = wred_sum(sum);
    const float invs = 1.0f / sum;
    for (int i = lane; i < HW_P; i += 32) S[row * HW_P + i] *= invs;
  }
  __syncthreads();

  // ---- in-place fp32 -> f16 of P (register-staged, barrier-safe) ----
  float tmp[46];                             // 16*736 = 256*46
  {
    const int b = threadIdx.x * 46;
#pragma unroll
    for (int i = 0; i < 46; ++i) tmp[i] = S[b + i];
  }
  __syncthreads();
  _Float16* Pf = (_Float16*)S;
  {
    const int b = threadIdx.x * 46;
#pragma unroll
    for (int i = 0; i < 46; ++i) Pf[b + i] = (_Float16)tmp[i];
  }
  __syncthreads();

  // ---- phase 3: O = P @ V  (P A-frags from LDS, V B-frags contiguous via Vt) ----
  const _Float16* Vb = Vt + (size_t)(t * H_N + h) * HD * HW_P;
  for (int j = wave; j < HD / 16; j += 8) {
    const int n0 = j * 16;
    v8f c = zero;
    for (int kt = 0; kt < HW_P / 32; ++kt) {
      const int k0 = kt * 32;
      const _Float16* pa = Pf + nl * HW_P + k0;
      v16h a = cat16(*(const v8h*)(pa + g * 16), *(const v8h*)(pa + g * 16 + 8));
      const _Float16* pb = Vb + (size_t)(n0 + nl) * HW_P + k0 + g * 16;
      c = wmma_f16(a, cat16(*(const v8h*)pb, *(const v8h*)(pb + 8)), c);
    }
#pragma unroll
    for (int r = 0; r < 8; ++r) {
      sampled[((size_t)t * M_TRK + m0 + g * 8 + r) * D_M + h * HD + n0 + nl] = (_Float16)c[r];
    }
  }
}

extern "C" void kernel_launch(void* const* d_in, const int* in_sizes, int n_in,
                              void* d_out, int out_size, void* d_ws, size_t ws_size,
                              hipStream_t stream) {
  (void)in_sizes; (void)n_in; (void)out_size; (void)ws_size;
  const float* features = (const float*)d_in[0];
  const float* tracks   = (const float*)d_in[1];
  const float* tok      = (const float*)d_in[2];
  // d_in[3] feature_positions recomputed analytically: (n%27, n/27)
  const float* W_q      = (const float*)d_in[4];
  const float* W_k      = (const float*)d_in[5];
  const float* g_q      = (const float*)d_in[6];
  const float* g_k      = (const float*)d_in[7];
  const float* W_out    = (const float*)d_in[8];
  float* out = (float*)d_out;

  char* ws = (char*)d_ws;
  size_t off = 0;
  auto alloc = [&](size_t bytes) -> char* {
    char* p = ws + off;
    off += (bytes + 255) & ~(size_t)255;
    return p;
  };
  const size_t nTok  = (size_t)T_DIM * M_TRK * D_M;   // 9,437,184
  const size_t nFeat = (size_t)T_DIM * HW_N * D_M;    // 13,436,928
  const size_t nW    = (size_t)D_M * D_M;

  _Float16* tokF  = (_Float16*)alloc(nTok * 2);
  _Float16* featF = (_Float16*)alloc(nFeat * 2);
  _Float16* WqF   = (_Float16*)alloc(nW * 2);
  _Float16* WkF   = (_Float16*)alloc(nW * 2);
  _Float16* WoF   = (_Float16*)alloc(nW * 2);
  float*    Qraw  = (float*)alloc(nTok * 4);          // later aliased by sampled f16
  float*    Kraw  = (float*)alloc(nFeat * 4);
  const size_t szQh = (size_t)T_DIM * H_N * M_TRK * HD_P * 2;
  const size_t szKh = (size_t)T_DIM * H_N * HW_P  * HD_P * 2;
  const size_t szVt = (size_t)T_DIM * H_N * HD    * HW_P * 2;
  _Float16* Qh = (_Float16*)alloc(szQh);
  _Float16* Kh = (_Float16*)alloc(szKh);
  _Float16* Vt = (_Float16*)alloc(szVt);
  _Float16* sampledF = (_Float16*)Qraw;               // alias: Qraw dead after k_ln_q

  // 1) precision conversion
  k_f32_to_f16<<<(int)((nTok  + 255) / 256), 256, 0, stream>>>(tok, tokF, (int)nTok);
  k_f32_to_f16<<<(int)((nFeat + 255) / 256), 256, 0, stream>>>(features, featF, (int)nFeat);
  k_f32_to_f16<<<(int)((nW + 255) / 256), 256, 0, stream>>>(W_q, WqF, (int)nW);
  k_f32_to_f16<<<(int)((nW + 255) / 256), 256, 0, stream>>>(W_k, WkF, (int)nW);
  k_f32_to_f16<<<(int)((nW + 255) / 256), 256, 0, stream>>>(W_out, WoF, (int)nW);

  // 2) zero padded Qh/Kh/Vt (contiguous span)
  k_zero<<<4096, 256, 0, stream>>>((uint4*)Qh, (szQh + szKh + szVt) / 16);

  // 3) projections (register-blocked WMMA GEMM): block tile 256x64
  k_gemm_f16f16_f32<<<dim3(18, 32), 256, 0, stream>>>(tokF,  WqF, Qraw, 8192,  1152, 1152);
  k_gemm_f16f16_f32<<<dim3(18, 46), 256, 0, stream>>>(featF, WkF, Kraw, 11664, 1152, 1152);

  // 4) RoPE / LayerNorm / head-split pack
  k_ln_q     <<<1024, 256, 0, stream>>>(Qraw, g_q, Qh);
  k_rope_ln_k<<<1458, 256, 0, stream>>>(Kraw, g_k, Kh);
  k_rope_v   <<<1458, 256, 0, stream>>>(features, Vt);

  // 5) attention (scores + bias + softmax + P@V), scores stay in LDS
  k_attn<<<dim3(32, 8, 16), 256, 0, stream>>>(Qh, Kh, Vt, tracks, sampledF);

  // 6) output projection (WMMA GEMM, fp32 out)
  k_gemm_f16f16_f32<<<dim3(18, 32), 256, 0, stream>>>(sampledF, WoF, out, 8192, 1152, 1152);
}
